// AtomCrossAttEncoder_86492051407609
// MI455X (gfx1250) — compile-verified, matrix operand order checked by reading, not run on hardware
//
#include <hip/hip_runtime.h>

#define TT 768
#define DD 24
#define SS 576
#define QQ 32
#define KK 128
#define CA 128
#define CP 16
#define CT 384
#define HH 4
#define DHH 32

typedef __attribute__((ext_vector_type(16))) __bf16 v16bf;
typedef __attribute__((ext_vector_type(8)))  float  v8f;
typedef __attribute__((ext_vector_type(2)))  float  v2f;

__device__ __forceinline__ __bf16 tob(float x) { return (__bf16)x; }

// ---------------------------------------------------------------------------
// 1. Per-atom conditioning: act (T*D, CA)
// ---------------------------------------------------------------------------
__global__ void k_atom_cond(const float* __restrict__ pos, const float* __restrict__ mask,
                            const float* __restrict__ charge, const int* __restrict__ elem,
                            const int* __restrict__ chars,
                            const float* __restrict__ wpos, const float* __restrict__ wmask,
                            const float* __restrict__ welem, const float* __restrict__ wchg,
                            const float* __restrict__ wname, float* __restrict__ act)
{
    int atom = blockIdx.x * 2 + (threadIdx.x >> 7);
    int c = threadIdx.x & 127;
    float p0 = pos[atom * 3 + 0], p1 = pos[atom * 3 + 1], p2 = pos[atom * 3 + 2];
    float mk = mask[atom];
    float a = p0 * wpos[c] + p1 * wpos[CA + c] + p2 * wpos[2 * CA + c]
            + mk * wmask[c]
            + welem[elem[atom] * CA + c]
            + asinhf(charge[atom]) * wchg[c];
#pragma unroll
    for (int i = 0; i < 4; ++i)
        a += wname[(i * 64 + chars[atom * 4 + i]) * CA + c];
    act[(size_t)atom * CA + c] = a * mk;
}

// ---------------------------------------------------------------------------
// 2. Gather queries (atoms -> S,Q layout)
// ---------------------------------------------------------------------------
__global__ void k_gather_q(const int* __restrict__ t2q, const float* __restrict__ act,
                           const float* __restrict__ amask, const int* __restrict__ uid,
                           const float* __restrict__ pos,
                           float* __restrict__ qcond, float* __restrict__ qmask,
                           int* __restrict__ quid, float* __restrict__ qrp)
{
    int row = blockIdx.x * 2 + (threadIdx.x >> 7);
    int c = threadIdx.x & 127;
    int idx = t2q[row];
    qcond[(size_t)row * CA + c] = act[(size_t)idx * CA + c];
    if (c == 0) { qmask[row] = amask[idx]; quid[row] = uid[idx]; }
    if (c < 3)  { qrp[row * 3 + c] = pos[idx * 3 + c]; }
}

// ---------------------------------------------------------------------------
// 3. Gather keys (S,Q space -> S,K layout)
// ---------------------------------------------------------------------------
__global__ void k_gather_k(const int* __restrict__ q2k, const float* __restrict__ qcond,
                           const float* __restrict__ qmask, const int* __restrict__ quid,
                           const float* __restrict__ qrp,
                           float* __restrict__ kcond, float* __restrict__ kmask,
                           int* __restrict__ kuid, float* __restrict__ krp)
{
    int row = blockIdx.x * 2 + (threadIdx.x >> 7);
    int c = threadIdx.x & 127;
    int idx = q2k[row];
    kcond[(size_t)row * CA + c] = qcond[(size_t)idx * CA + c];
    if (c == 0) { kmask[row] = qmask[idx]; kuid[row] = quid[idx]; }
    if (c < 3)  { krp[row * 3 + c] = qrp[idx * 3 + c]; }
}

// ---------------------------------------------------------------------------
// 4. Generic GEMM: Y[M,N] = (relu?)(X[M,128..]) @ W[Kd,N], bf16 WMMA, f32 acc
//    One wave per 16x16 output tile; 4 waves / block.
// ---------------------------------------------------------------------------
__global__ void k_gemm_bf16(const float* __restrict__ X, const float* __restrict__ W,
                            float* __restrict__ Y, int M, int N, int Kd, int relu)
{
    int wave = threadIdx.x >> 5, lane = threadIdx.x & 31;
    int ntiles = N >> 4;
    int tile = blockIdx.x * 4 + wave;
    if (tile >= (M >> 4) * ntiles) return;
    int mt = tile / ntiles, nt = tile % ntiles;
    int lo = lane & 15, kb = (lane >> 4) * 8;
    int row = mt * 16 + lo;
    int col = nt * 16 + lo;
    v8f acc = {};
    for (int kc = 0; kc < Kd; kc += 32) {
        const float* xp = X + (size_t)row * Kd + kc + kb;
        float xa[16];
        *reinterpret_cast<float4*>(&xa[0])  = *reinterpret_cast<const float4*>(xp);
        *reinterpret_cast<float4*>(&xa[4])  = *reinterpret_cast<const float4*>(xp + 4);
        *reinterpret_cast<float4*>(&xa[8])  = *reinterpret_cast<const float4*>(xp + 16);
        *reinterpret_cast<float4*>(&xa[12]) = *reinterpret_cast<const float4*>(xp + 20);
        if (relu) {
#pragma unroll
            for (int i = 0; i < 16; ++i) xa[i] = fmaxf(xa[i], 0.f);
        }
        v16bf a;
#pragma unroll
        for (int i = 0; i < 16; ++i) a[i] = tob(xa[i]);
        v16bf b;
        const float* wp = W + (size_t)(kc + kb) * N + col;
#pragma unroll
        for (int j = 0; j < 8; ++j) {
            b[j]     = tob(wp[(size_t)j * N]);
            b[8 + j] = tob(wp[(size_t)(16 + j) * N]);
        }
        acc = __builtin_amdgcn_wmma_f32_16x16x32_bf16(false, a, false, b, (short)0, acc, false, false);
    }
    int hi8 = (lane >= 16) ? 8 : 0;
#pragma unroll
    for (int r = 0; r < 8; ++r)
        Y[(size_t)(mt * 16 + r + hi8) * N + col] = acc[r];
}

// ---------------------------------------------------------------------------
// 5. Fused pair kernel: base pair + 3x (16x16) MLP via f32 WMMA + bias
//    One wave handles 16 rows (one (s,q), 16 consecutive keys).
// ---------------------------------------------------------------------------
__global__ void __launch_bounds__(256) k_pair(
    const float* __restrict__ row1, const float* __restrict__ col1,
    const float* __restrict__ qrp, const float* __restrict__ krp,
    const int* __restrict__ quid, const int* __restrict__ kuid,
    const float* __restrict__ wpo, const float* __restrict__ wpd,
    const float* __restrict__ wpv, const float* __restrict__ w1,
    const float* __restrict__ w2, const float* __restrict__ w3,
    const float* __restrict__ wb, float* __restrict__ out_pair,
    float* __restrict__ biasws)
{
    __shared__ float lds_base[8][256];
    __shared__ float lds_t[8][256];
    __shared__ float lds_fin[8][256];
    int wave = threadIdx.x >> 5, lane = threadIdx.x & 31;
    int gw = blockIdx.x * 8 + wave;
    int sq = gw >> 3;           // (s*Q + q)
    int k0 = (gw & 7) << 4;     // key tile base
    int s  = sq >> 5;           // Q == 32
    int m  = lane & 15;
    int off = (lane >> 4) * 2;
    int n = m;                  // B/D fragment column
    int hi8 = (lane >= 16) ? 8 : 0;
    int key = s * KK + k0 + m;

    float ox = qrp[sq * 3 + 0] - krp[key * 3 + 0];
    float oy = qrp[sq * 3 + 1] - krp[key * 3 + 1];
    float oz = qrp[sq * 3 + 2] - krp[key * 3 + 2];
    float invd = 1.f / (1.f + ox * ox + oy * oy + oz * oz);
    float vf = (quid[sq] == kuid[key]) ? 1.f : 0.f;

    // base pair, per-lane in A-fragment channel order
    float pb[8];
#pragma unroll
    for (int c = 0; c < 4; ++c)
#pragma unroll
        for (int t = 0; t < 2; ++t) {
            int ch = c * 4 + off + t;
            float x = row1[(size_t)sq * CP + ch] + col1[(size_t)key * CP + ch]
                    + vf * (ox * wpo[ch] + oy * wpo[CP + ch] + oz * wpo[2 * CP + ch]
                            + invd * wpd[ch] + wpv[ch]);
            pb[c * 2 + t] = x;
            lds_base[wave][m * 16 + ch] = x;
        }
    __syncthreads();

    // stage 1: t1 = relu(pair) @ w1
    v8f d = {};
#pragma unroll
    for (int c = 0; c < 4; ++c) {
        v2f a = { fmaxf(pb[2 * c], 0.f), fmaxf(pb[2 * c + 1], 0.f) };
        v2f b = { w1[(c * 4 + off) * CP + n], w1[(c * 4 + off + 1) * CP + n] };
        d = __builtin_amdgcn_wmma_f32_16x16x4_f32(false, a, false, b, (short)0, d, false, false);
    }
#pragma unroll
    for (int r = 0; r < 8; ++r) lds_t[wave][(r + hi8) * 16 + n] = fmaxf(d[r], 0.f);
    __syncthreads();
    float u[8];
#pragma unroll
    for (int c = 0; c < 4; ++c) {
        u[2 * c]     = lds_t[wave][m * 16 + c * 4 + off];
        u[2 * c + 1] = lds_t[wave][m * 16 + c * 4 + off + 1];
    }

    // stage 2
    d = {};
#pragma unroll
    for (int c = 0; c < 4; ++c) {
        v2f a = { u[2 * c], u[2 * c + 1] };
        v2f b = { w2[(c * 4 + off) * CP + n], w2[(c * 4 + off + 1) * CP + n] };
        d = __builtin_amdgcn_wmma_f32_16x16x4_f32(false, a, false, b, (short)0, d, false, false);
    }
    __syncthreads();
#pragma unroll
    for (int r = 0; r < 8; ++r) lds_t[wave][(r + hi8) * 16 + n] = fmaxf(d[r], 0.f);
    __syncthreads();
#pragma unroll
    for (int c = 0; c < 4; ++c) {
        u[2 * c]     = lds_t[wave][m * 16 + c * 4 + off];
        u[2 * c + 1] = lds_t[wave][m * 16 + c * 4 + off + 1];
    }

    // stage 3
    d = {};
#pragma unroll
    for (int c = 0; c < 4; ++c) {
        v2f a = { u[2 * c], u[2 * c + 1] };
        v2f b = { w3[(c * 4 + off) * CP + n], w3[(c * 4 + off + 1) * CP + n] };
        d = __builtin_amdgcn_wmma_f32_16x16x4_f32(false, a, false, b, (short)0, d, false, false);
    }

    // final pair = base + t3; write out + keep for bias
#pragma unroll
    for (int r = 0; r < 8; ++r) {
        int mm = r + hi8;
        float fin = lds_base[wave][mm * 16 + n] + d[r];
        lds_fin[wave][mm * 16 + n] = fin;
        out_pair[((size_t)sq * KK + k0 + mm) * CP + n] = fin;
    }
    __syncthreads();

    // bias = pair @ wb (16 -> 4); each lane does 2 heads for its row
    int h0 = (lane >> 4) * 2;
    float b0 = 0.f, b1 = 0.f;
#pragma unroll
    for (int nn = 0; nn < 16; ++nn) {
        float p = lds_fin[wave][m * 16 + nn];
        b0 += p * wb[nn * HH + h0];
        b1 += p * wb[nn * HH + h0 + 1];
    }
    size_t bi = ((size_t)sq * KK + k0 + m) * HH;
    biasws[bi + h0]     = b0;
    biasws[bi + h0 + 1] = b1;
}

// ---------------------------------------------------------------------------
// 6. Fused attention per subset: 4 waves = 4 heads, bf16 WMMA + reg softmax
// ---------------------------------------------------------------------------
__global__ void __launch_bounds__(128) k_attn(
    const float* __restrict__ qws, const float* __restrict__ kws, const float* __restrict__ vws,
    const float* __restrict__ biasws, const float* __restrict__ qcond,
    const float* __restrict__ qmask, const float* __restrict__ kmask,
    const float* __restrict__ wo, float* __restrict__ out_skip)
{
    __shared__ __bf16 p_lds[HH][16 * KK];
    __shared__ __bf16 o_lds[QQ * CA];
    int s = blockIdx.x;
    int h = threadIdx.x >> 5;
    int lane = threadIdx.x & 31;
    int lo = lane & 15;
    int kb = (lane >> 4) * 8;
    int hi8 = (lane >= 16) ? 8 : 0;
    const float rscale = 0.1767766952966369f; // 1/sqrt(32)

    for (int Mt = 0; Mt < 2; ++Mt) {
        // logits strip (16 q-rows x 128 keys) = q_h @ k_h^T
        v16bf aq;
        {
            const float* qp = qws + ((size_t)s * QQ + Mt * 16 + lo) * (HH * DHH) + h * DHH;
#pragma unroll
            for (int j = 0; j < 8; ++j) { aq[j] = tob(qp[kb + j]); aq[8 + j] = tob(qp[kb + 16 + j]); }
        }
        v8f dacc[8];
#pragma unroll
        for (int nt = 0; nt < 8; ++nt) {
            v16bf bk;
            const float* kp = kws + ((size_t)s * KK + nt * 16 + lo) * (HH * DHH) + h * DHH;
#pragma unroll
            for (int j = 0; j < 8; ++j) { bk[j] = tob(kp[kb + j]); bk[8 + j] = tob(kp[kb + 16 + j]); }
            v8f z = {};
            dacc[nt] = __builtin_amdgcn_wmma_f32_16x16x32_bf16(false, aq, false, bk, (short)0, z, false, false);
        }
        // scale + bias + mask + row softmax (rows live across 16-lane halves)
#pragma unroll
        for (int r = 0; r < 8; ++r) {
            int qrow = Mt * 16 + r + hi8;
            float mr = -3.0e38f;
#pragma unroll
            for (int nt = 0; nt < 8; ++nt) {
                int keyi = nt * 16 + lo;
                float l = dacc[nt][r] * rscale
                        + biasws[(((size_t)s * QQ + qrow) * KK + keyi) * HH + h];
                l = (kmask[s * KK + keyi] > 0.f) ? l : -1.0e9f;
                dacc[nt][r] = l;
                mr = fmaxf(mr, l);
            }
            mr = fmaxf(mr, __shfl_xor(mr, 1));
            mr = fmaxf(mr, __shfl_xor(mr, 2));
            mr = fmaxf(mr, __shfl_xor(mr, 4));
            mr = fmaxf(mr, __shfl_xor(mr, 8));
            float sr = 0.f;
#pragma unroll
            for (int nt = 0; nt < 8; ++nt) {
                float e = __expf(dacc[nt][r] - mr);
                dacc[nt][r] = e; sr += e;
            }
            sr += __shfl_xor(sr, 1);
            sr += __shfl_xor(sr, 2);
            sr += __shfl_xor(sr, 4);
            sr += __shfl_xor(sr, 8);
            float inv = 1.f / sr;
#pragma unroll
            for (int nt = 0; nt < 8; ++nt)
                p_lds[h][(r + hi8) * KK + nt * 16 + lo] = tob(dacc[nt][r] * inv);
        }
        __syncthreads();
        // o strip = attn @ v_h
#pragma unroll
        for (int ntv = 0; ntv < 2; ++ntv) {
            v8f oacc = {};
#pragma unroll
            for (int kc = 0; kc < 4; ++kc) {
                v16bf ap;
                const __bf16* pp = &p_lds[h][lo * KK + kc * 32];
#pragma unroll
                for (int j = 0; j < 8; ++j) { ap[j] = pp[kb + j]; ap[8 + j] = pp[kb + 16 + j]; }
                v16bf bv;
                const float* vp = vws + ((size_t)s * KK + kc * 32) * (HH * DHH) + h * DHH + ntv * 16 + lo;
#pragma unroll
                for (int j = 0; j < 8; ++j) {
                    bv[j]     = tob(vp[(size_t)(kb + j) * (HH * DHH)]);
                    bv[8 + j] = tob(vp[(size_t)(kb + 16 + j) * (HH * DHH)]);
                }
                oacc = __builtin_amdgcn_wmma_f32_16x16x32_bf16(false, ap, false, bv, (short)0, oacc, false, false);
            }
#pragma unroll
            for (int r = 0; r < 8; ++r)
                o_lds[(Mt * 16 + r + hi8) * CA + h * DHH + ntv * 16 + lo] = tob(oacc[r]);
        }
        __syncthreads();
    }
    // skip = (qcond + o @ wo) * qmask
#pragma unroll
    for (int mt2 = 0; mt2 < 2; ++mt2)
#pragma unroll
        for (int ntl = 0; ntl < 2; ++ntl) {
            int nt = h * 2 + ntl;
            v8f acc = {};
#pragma unroll
            for (int kc = 0; kc < 4; ++kc) {
                v16bf ao;
                const __bf16* op = &o_lds[(mt2 * 16 + lo) * CA + kc * 32];
#pragma unroll
                for (int j = 0; j < 8; ++j) { ao[j] = op[kb + j]; ao[8 + j] = op[kb + 16 + j]; }
                v16bf bw;
                const float* wp = wo + (size_t)(kc * 32 + kb) * CA + nt * 16 + lo;
#pragma unroll
                for (int j = 0; j < 8; ++j) {
                    bw[j]     = tob(wp[(size_t)j * CA]);
                    bw[8 + j] = tob(wp[(size_t)(16 + j) * CA]);
                }
                acc = __builtin_amdgcn_wmma_f32_16x16x32_bf16(false, ao, false, bw, (short)0, acc, false, false);
            }
#pragma unroll
            for (int r = 0; r < 8; ++r) {
                size_t qi = (size_t)s * QQ + mt2 * 16 + r + hi8;
                int n = nt * 16 + lo;
                float v = (qcond[qi * CA + n] + acc[r]) * qmask[qi];
                out_skip[qi * CA + n] = v;
            }
        }
}

// ---------------------------------------------------------------------------
// 7. Token pooling: gather proj -> (T,D,CT), relu, mask-mean over D
// ---------------------------------------------------------------------------
__global__ void k_token(const int* __restrict__ q2t, const float* __restrict__ amask,
                        const float* __restrict__ proj, float* __restrict__ out_token)
{
    int t = blockIdx.x, c = threadIdx.x;
    float num = 0.f, den = 0.f;
    for (int d = 0; d < DD; ++d) {
        float mk = amask[t * DD + d];
        int idx = q2t[t * DD + d];
        num += mk * fmaxf(proj[(size_t)idx * CT + c], 0.f);
        den += mk;
    }
    out_token[(size_t)t * CT + c] = num / (den + 1e-10f);
}

// ---------------------------------------------------------------------------
extern "C" void kernel_launch(void* const* d_in, const int* in_sizes, int n_in,
                              void* d_out, int out_size, void* d_ws, size_t ws_size,
                              hipStream_t stream)
{
    (void)in_sizes; (void)n_in; (void)out_size; (void)ws_size;
    const float* ref_positions = (const float*)d_in[0];
    const float* ref_mask      = (const float*)d_in[1];
    const float* ref_charge    = (const float*)d_in[2];
    const float* atom_mask     = (const float*)d_in[3];
    const int*   ref_element   = (const int*)d_in[4];
    const int*   ref_name      = (const int*)d_in[5];
    const int*   ref_uid       = (const int*)d_in[6];
    const int*   t2q           = (const int*)d_in[7];
    const int*   q2k           = (const int*)d_in[8];
    const int*   q2t           = (const int*)d_in[9];
    const float* w_ref_pos     = (const float*)d_in[10];
    const float* w_ref_mask    = (const float*)d_in[11];
    const float* w_ref_element = (const float*)d_in[12];
    const float* w_ref_charge  = (const float*)d_in[13];
    const float* w_ref_name    = (const float*)d_in[14];
    const float* w_s2p_row1    = (const float*)d_in[15];
    const float* w_s2p_col1    = (const float*)d_in[16];
    const float* w_pair_off1   = (const float*)d_in[17];
    const float* w_pair_dist1  = (const float*)d_in[18];
    const float* w_pair_valid  = (const float*)d_in[19];
    const float* w_mlp1        = (const float*)d_in[20];
    const float* w_mlp2        = (const float*)d_in[21];
    const float* w_mlp3        = (const float*)d_in[22];
    const float* wq            = (const float*)d_in[23];
    const float* wk            = (const float*)d_in[24];
    const float* wv            = (const float*)d_in[25];
    const float* wb            = (const float*)d_in[26];
    const float* wo            = (const float*)d_in[27];
    const float* w_proj        = (const float*)d_in[28];

    float* out = (float*)d_out;
    // output slice offsets (floats), in reference return order
    const size_t OUT_TOKEN = 0;
    const size_t OUT_SKIP  = OUT_TOKEN + (size_t)TT * CT;
    const size_t OUT_QMASK = OUT_SKIP  + (size_t)SS * QQ * CA;
    const size_t OUT_QCOND = OUT_QMASK + (size_t)SS * QQ;
    const size_t OUT_KMASK = OUT_QCOND + (size_t)SS * QQ * CA;
    const size_t OUT_KCOND = OUT_KMASK + (size_t)SS * KK;
    const size_t OUT_PAIR  = OUT_KCOND + (size_t)SS * KK * CA;

    // workspace layout (floats)
    float* ws = (float*)d_ws;
    size_t o = 0;
    float* ws_act  = ws + o; o += (size_t)TT * DD * CA;
    float* ws_qrp  = ws + o; o += (size_t)SS * QQ * 3;
    float* ws_krp  = ws + o; o += (size_t)SS * KK * 3;
    int*   ws_quid = (int*)(ws + o); o += (size_t)SS * QQ;
    int*   ws_kuid = (int*)(ws + o); o += (size_t)SS * KK;
    float* ws_row1 = ws + o; o += (size_t)SS * QQ * CP;
    float* ws_col1 = ws + o; o += (size_t)SS * KK * CP;
    float* ws_q    = ws + o; o += (size_t)SS * QQ * HH * DHH;
    float* ws_k    = ws + o; o += (size_t)SS * KK * HH * DHH;
    float* ws_v    = ws + o; o += (size_t)SS * KK * HH * DHH;
    float* ws_bias = ws + o; o += (size_t)SS * QQ * KK * HH;
    float* ws_proj = ws + o; o += (size_t)SS * QQ * CT;

    // 1. per-atom conditioning
    k_atom_cond<<<(TT * DD) / 2, 256, 0, stream>>>(
        ref_positions, ref_mask, ref_charge, ref_element, ref_name,
        w_ref_pos, w_ref_mask, w_ref_element, w_ref_charge, w_ref_name, ws_act);

    // 2. gather queries
    k_gather_q<<<(SS * QQ) / 2, 256, 0, stream>>>(
        t2q, ws_act, atom_mask, ref_uid, ref_positions,
        out + OUT_QCOND, out + OUT_QMASK, ws_quid, ws_qrp);

    // 3. gather keys
    k_gather_k<<<(SS * KK) / 2, 256, 0, stream>>>(
        q2k, out + OUT_QCOND, out + OUT_QMASK, ws_quid, ws_qrp,
        out + OUT_KCOND, out + OUT_KMASK, ws_kuid, ws_krp);

    // 4. row1 / col1 (relu GEMM, 128->16)
    k_gemm_bf16<<<((SS * QQ / 16) * 1 + 3) / 4, 128, 0, stream>>>(
        out + OUT_QCOND, w_s2p_row1, ws_row1, SS * QQ, CP, CA, 1);
    k_gemm_bf16<<<((SS * KK / 16) * 1 + 3) / 4, 128, 0, stream>>>(
        out + OUT_KCOND, w_s2p_col1, ws_col1, SS * KK, CP, CA, 1);

    // 5. q / k / v projections (128->128)
    k_gemm_bf16<<<((SS * QQ / 16) * 8 + 3) / 4, 128, 0, stream>>>(
        out + OUT_QCOND, wq, ws_q, SS * QQ, HH * DHH, CA, 0);
    k_gemm_bf16<<<((SS * KK / 16) * 8 + 3) / 4, 128, 0, stream>>>(
        out + OUT_KCOND, wk, ws_k, SS * KK, HH * DHH, CA, 0);
    k_gemm_bf16<<<((SS * KK / 16) * 8 + 3) / 4, 128, 0, stream>>>(
        out + OUT_KCOND, wv, ws_v, SS * KK, HH * DHH, CA, 0);

    // 6. fused pair + MLP + bias (dominant memory cost, single pass)
    k_pair<<<(SS * QQ * KK / 16) / 8, 256, 0, stream>>>(
        ws_row1, ws_col1, ws_qrp, ws_krp, ws_quid, ws_kuid,
        w_pair_off1, w_pair_dist1, w_pair_valid, w_mlp1, w_mlp2, w_mlp3,
        wb, out + OUT_PAIR, ws_bias);

    // 7. fused attention + output projection + residual + mask -> skip
    k_attn<<<SS, 128, 0, stream>>>(
        ws_q, ws_k, ws_v, ws_bias, out + OUT_QCOND,
        out + OUT_QMASK, out + OUT_KMASK, wo, out + OUT_SKIP);

    // 8. proj = skip @ w_proj (128->384)
    k_gemm_bf16<<<((SS * QQ / 16) * (CT / 16) + 3) / 4, 128, 0, stream>>>(
        out + OUT_SKIP, w_proj, ws_proj, SS * QQ, CT, CA, 0);

    // 9. token pooling
    k_token<<<TT, CT, 0, stream>>>(q2t, atom_mask, ws_proj, out + OUT_TOKEN);
}
